// Unidepth_Decoder_71390946394902
// MI455X (gfx1250) — compile-verified, matrix operand order checked by reading, not run on hardware
//
#include <hip/hip_runtime.h>
#include <math.h>

// ---------------------------------------------------------------------------
// UniDepth decoder for MI455X (gfx1250).
// Big GEMMs (in_features MLP, kv projections) run as bf16 WMMA
// (v_wmma_f32_16x16x32_bf16) with double-buffered ASYNC global->LDS staging
// (global_load_async_to_lds_b128 / s_wait_asynccnt). Tiny cls-path (16 rows)
// runs as VALU kernels.
// ---------------------------------------------------------------------------

typedef __bf16 bf16;
typedef __bf16 v16bf __attribute__((ext_vector_type(16)));
typedef float  v8f   __attribute__((ext_vector_type(8)));

#define DEV static __device__ __forceinline__

DEV float gelu_exact(float x) {
  return 0.5f * x * (1.0f + erff(x * 0.70710678118654752f));
}

// 256-thread block reductions (red = 256-float LDS scratch)
DEV float block_reduce_sum(float v, float* red) {
  int tid = threadIdx.x;
  red[tid] = v; __syncthreads();
  for (int off = 128; off > 0; off >>= 1) {
    if (tid < off) red[tid] += red[tid + off];
    __syncthreads();
  }
  float r = red[0]; __syncthreads();
  return r;
}
DEV float block_reduce_max(float v, float* red) {
  int tid = threadIdx.x;
  red[tid] = v; __syncthreads();
  for (int off = 128; off > 0; off >>= 1) {
    if (tid < off) red[tid] = fmaxf(red[tid], red[tid + off]);
    __syncthreads();
  }
  float r = red[0]; __syncthreads();
  return r;
}

DEV void async_fill_b128(unsigned lds_off, const void* gaddr) {
  // CDNA5 async copy: memory -> LDS, tracked with ASYNCcnt (no VGPR round trip)
  asm volatile("global_load_async_to_lds_b128 %0, %1, off"
               :: "v"(lds_off), "v"(gaddr) : "memory");
}
DEV void wait_async0() {
  asm volatile("s_wait_asynccnt 0x0" ::: "memory");
}

// ---------------------------------------------------------------------------
// WMMA GEMM:  C[M x N] = act( A_bf16[M x K] @ Bt_bf16[N x K]^T + bias )
// 256 threads = 8 waves. WG tile 128x128, K staged 64 at a time via LDS,
// double-buffered with async global->LDS fills overlapping the WMMAs.
// Wave (wm,wn): rows [wm*64,+64), cols [wn*32,+32) => 4x2 accumulators.
// flags: bit0 = GELU, bit1 = store bf16 (else f32)
// ---------------------------------------------------------------------------
__global__ __launch_bounds__(256)
void wmma_gemm_bf16(const bf16* __restrict__ A, const bf16* __restrict__ Bt,
                    const float* __restrict__ bias, void* __restrict__ C,
                    int M, int N, int K, int flags)
{
  __shared__ __align__(16) bf16 lA[2][128][72];
  __shared__ __align__(16) bf16 lB[2][128][72];

  const int tid  = threadIdx.x;
  const int lane = tid & 31;
  const int wave = tid >> 5;
  const int wm   = wave & 1;     // row half of WG tile
  const int wn   = wave >> 1;    // col quarter of WG tile
  const int half = lane >> 4;    // 0/1 (lane group)
  const int lr   = lane & 15;
  const int rowBase = blockIdx.y * 128;
  const int colBase = blockIdx.x * 128;

  v8f acc[4][2];
  const v8f vzero = {0.f,0.f,0.f,0.f,0.f,0.f,0.f,0.f};
#pragma unroll
  for (int i = 0; i < 4; ++i)
#pragma unroll
    for (int j = 0; j < 2; ++j) acc[i][j] = vzero;

  // Issue one 128x64-half stage (16B per thread-chunk, 4 chunks each for A/B)
  auto stage = [&](int p, int k0) {
#pragma unroll
    for (int it = 0; it < 4; ++it) {
      int t = tid + it * 256;
      int r = t >> 3;
      int c = (t & 7) << 3;
      int gr = rowBase + r;
      if (gr < M) {
        async_fill_b128((unsigned)(size_t)&lA[p][r][c],
                        A + (size_t)gr * K + (k0 + c));
      } else {
        uint4 z; z.x = z.y = z.z = z.w = 0u;
        *(uint4*)&lA[p][r][c] = z;       // tail rows: zero via DS store
      }
      async_fill_b128((unsigned)(size_t)&lB[p][r][c],
                      Bt + (size_t)(colBase + r) * K + (k0 + c));
    }
  };

  const int nst = K >> 6;          // K is a multiple of 64
  stage(0, 0);
  wait_async0();
  __syncthreads();

  for (int s = 0; s < nst; ++s) {
    const int p = s & 1;
    if (s + 1 < nst) stage(p ^ 1, (s + 1) << 6);   // prefetch next stage

#pragma unroll
    for (int kk = 0; kk < 64; kk += 32) {
      // A-fragment (16-bit, 16x32): lanes 0-15 hold K {0..7,16..23},
      // lanes 16-31 hold K {8..15,24..31} for row = lr.
      const int kb = kk + half * 8;
      v16bf af[4];
      v16bf bfv[2];
#pragma unroll
      for (int j = 0; j < 2; ++j) {
        int cidx = wn * 32 + j * 16 + lr;
        uint4* dst = (uint4*)&bfv[j];
        dst[0] = *(const uint4*)&lB[p][cidx][kb];
        dst[1] = *(const uint4*)&lB[p][cidx][kb + 16];
      }
#pragma unroll
      for (int i = 0; i < 4; ++i) {
        int r = wm * 64 + i * 16 + lr;
        uint4* dst = (uint4*)&af[i];
        dst[0] = *(const uint4*)&lA[p][r][kb];
        dst[1] = *(const uint4*)&lA[p][r][kb + 16];
      }
#pragma unroll
      for (int i = 0; i < 4; ++i)
#pragma unroll
        for (int j = 0; j < 2; ++j)
          acc[i][j] = __builtin_amdgcn_wmma_f32_16x16x32_bf16(
              false, af[i], false, bfv[j], (short)0, acc[i][j], false, false);
    }

    wait_async0();      // next-stage fills (issued above) are in LDS
    __syncthreads();    // all waves done reading buffer p + all fills visible
  }

  // ---- epilogue: C layout VGPR v -> row = v + 8*half, col = lr ----
  const bool outbf = (flags & 2) != 0;
  const bool dog   = (flags & 1) != 0;
#pragma unroll
  for (int i = 0; i < 4; ++i) {
#pragma unroll
    for (int j = 0; j < 2; ++j) {
      int col = colBase + wn * 32 + j * 16 + lr;
      float bv = bias ? bias[col] : 0.0f;
#pragma unroll
      for (int v = 0; v < 8; ++v) {
        int row = rowBase + wm * 64 + i * 16 + half * 8 + v;
        if (row < M) {
          float x = acc[i][j][v] + bv;
          if (dog) x = gelu_exact(x);
          if (outbf) ((bf16*)C)[(size_t)row * N + col] = (bf16)x;
          else       ((float*)C)[(size_t)row * N + col] = x;
        }
      }
    }
  }
}

// ---------------------------------------------------------------------------
// Weight prep: Wt_bf16[N x K] = (bf16) W_f32[K x N]^T   (K,N multiples of 32)
// ---------------------------------------------------------------------------
__global__ __launch_bounds__(256)
void transpose_w_bf16(const float* __restrict__ W, bf16* __restrict__ Wt,
                      int K, int N)
{
  __shared__ float tile[32][33];
  const int tid = threadIdx.x;
  const int tx = tid & 31;
  const int ty = tid >> 5;           // 0..7
  const int nb = blockIdx.x * 32;
  const int kb = blockIdx.y * 32;
#pragma unroll
  for (int i = 0; i < 4; ++i) {
    int k = ty + i * 8;
    tile[k][tx] = W[(size_t)(kb + k) * N + nb + tx];
  }
  __syncthreads();
#pragma unroll
  for (int i = 0; i < 4; ++i) {
    int n = ty + i * 8;
    Wt[(size_t)(nb + n) * K + kb + tx] = (bf16)tile[tx][n];
  }
}

// ---------------------------------------------------------------------------
// prep A for in_features MLP: gather moveaxis(features) + pos_embed, LN, bf16.
// One WG per row m in [0, 19200): b=m/4800, t=m%4800, r=t/1200, n=t%1200.
// ---------------------------------------------------------------------------
__global__ __launch_bounds__(256)
void prep_in_features(const float* __restrict__ features,
                      const float* __restrict__ pos,
                      const float* __restrict__ g, const float* __restrict__ bb,
                      bf16* __restrict__ A)
{
  __shared__ float red[256];
  const int m   = blockIdx.x;
  const int tid = threadIdx.x;
  const int bi  = m / 4800;
  const int t   = m % 4800;
  const int r   = t / 1200;
  const int nidx = t % 1200;
  const size_t fbase = ((size_t)(bi * 1200 + nidx)) * 1024u;
  const size_t pbase = ((size_t)bi * 4800 + t) * 1024u;

  float x[4]; float s = 0.f, sq = 0.f;
#pragma unroll
  for (int i = 0; i < 4; ++i) {
    int dd = tid + i * 256;
    float v = features[(fbase + dd) * 4 + r] + pos[pbase + dd];
    x[i] = v; s += v; sq += v * v;
  }
  float tot   = block_reduce_sum(s,  red);
  float totsq = block_reduce_sum(sq, red);
  float mean = tot * (1.0f / 1024.0f);
  float var  = totsq * (1.0f / 1024.0f) - mean * mean;
  float rstd = rsqrtf(var + 1e-5f);
#pragma unroll
  for (int i = 0; i < 4; ++i) {
    int dd = tid + i * 256;
    float y = (x[i] - mean) * rstd * g[dd] + bb[dd];
    A[(size_t)m * 1024 + dd] = (bf16)y;
  }
}

// ---------------------------------------------------------------------------
// prep A for kv GEMM: LN(ctx) in bf16, ctx row j<4800 -> feats, else cls0.
// One WG per row m in [0, 19216): b=m/4804, j=m%4804.
// ---------------------------------------------------------------------------
__global__ __launch_bounds__(256)
void prep_ctx(const float* __restrict__ feat, const float* __restrict__ cls0,
              const float* __restrict__ g, const float* __restrict__ bb,
              bf16* __restrict__ A)
{
  __shared__ float red[256];
  const int m   = blockIdx.x;
  const int tid = threadIdx.x;
  const int bi  = m / 4804;
  const int j   = m % 4804;
  const float* src = (j < 4800) ? (feat + ((size_t)bi * 4800 + j) * 1024)
                                : (cls0 + ((size_t)bi * 4 + (j - 4800)) * 1024);
  float x[4]; float s = 0.f, sq = 0.f;
#pragma unroll
  for (int i = 0; i < 4; ++i) {
    int dd = tid + i * 256;
    float v = src[dd];
    x[i] = v; s += v; sq += v * v;
  }
  float tot   = block_reduce_sum(s,  red);
  float totsq = block_reduce_sum(sq, red);
  float mean = tot * (1.0f / 1024.0f);
  float var  = totsq * (1.0f / 1024.0f) - mean * mean;
  float rstd = rsqrtf(var + 1e-5f);
#pragma unroll
  for (int i = 0; i < 4; ++i) {
    int dd = tid + i * 256;
    float y = (x[i] - mean) * rstd * g[dd] + bb[dd];
    A[(size_t)m * 1024 + dd] = (bf16)y;
  }
}

// ---------------------------------------------------------------------------
// Small LayerNorm over 1024 cols, fp32 out. One WG per row.
// ---------------------------------------------------------------------------
__global__ __launch_bounds__(256)
void ln_rows(const float* __restrict__ X, const float* __restrict__ g,
             const float* __restrict__ bb, float* __restrict__ Y)
{
  __shared__ float red[256];
  const int m   = blockIdx.x;
  const int tid = threadIdx.x;
  const float* src = X + (size_t)m * 1024;
  float x[4]; float s = 0.f, sq = 0.f;
#pragma unroll
  for (int i = 0; i < 4; ++i) {
    int dd = tid + i * 256;
    float v = src[dd];
    x[i] = v; s += v; sq += v * v;
  }
  float tot   = block_reduce_sum(s,  red);
  float totsq = block_reduce_sum(sq, red);
  float mean = tot * (1.0f / 1024.0f);
  float var  = totsq * (1.0f / 1024.0f) - mean * mean;
  float rstd = rsqrtf(var + 1e-5f);
#pragma unroll
  for (int i = 0; i < 4; ++i) {
    int dd = tid + i * 256;
    Y[(size_t)m * 1024 + dd] = (x[i] - mean) * rstd * g[dd] + bb[dd];
  }
}

// ---------------------------------------------------------------------------
// Small FC for the 16-row cls path:
//   out[row][o] = maybe(res + ls*) ( maybe_gelu( X[row] . W[:,o] + bias[o]
//                                               + lpos[row%4][o] ) )
// One WG per row. K <= 4096.  (out and res may alias: per-element RMW.)
// ---------------------------------------------------------------------------
__global__ __launch_bounds__(256)
void fc_small(const float* __restrict__ X, const float* __restrict__ W,
              const float* __restrict__ bias, float* out,
              int K, int N, int dogelu,
              const float* __restrict__ lpos, const float* res,
              const float* __restrict__ ls)
{
  __shared__ float lX[4096];
  const int row = blockIdx.x;
  const int tid = threadIdx.x;
  for (int k = tid; k < K; k += 256) lX[k] = X[(size_t)row * K + k];
  __syncthreads();
  for (int o = tid; o < N; o += 256) {
    float acc = bias ? bias[o] : 0.0f;
    for (int k = 0; k < K; ++k) acc += lX[k] * W[(size_t)k * N + o];
    if (lpos)   acc += lpos[(row & 3) * N + o];
    if (dogelu) acc = gelu_exact(acc);
    if (res)    acc = res[(size_t)row * N + o] + ls[o] * acc;
    out[(size_t)row * N + o] = acc;
  }
}

// ---------------------------------------------------------------------------
// Attention with 4 queries per batch over 4804 keys. One WG per (b, qi).
// KV rows: [k(0..1023) | v(1024..2047)], row index b*4804 + j.
// ---------------------------------------------------------------------------
__global__ __launch_bounds__(256)
void attn_small(const float* __restrict__ Q, const float* __restrict__ KV,
                float* __restrict__ O)
{
  __shared__ float lq[1024];
  __shared__ float lg[4804];
  __shared__ float red[256];
  const int tid = threadIdx.x;
  const int bi  = blockIdx.x >> 2;
  const int qi  = blockIdx.x & 3;
  const float* qrow = Q + ((size_t)bi * 4 + qi) * 1024;
  for (int k = tid; k < 1024; k += 256) lq[k] = qrow[k];
  __syncthreads();
  for (int j = tid; j < 4804; j += 256) {
    const float* krow = KV + ((size_t)bi * 4804 + j) * 2048;
    float s = 0.f;
    for (int k = 0; k < 1024; ++k) s += lq[k] * krow[k];
    lg[j] = s * 0.03125f;                 // 1/sqrt(1024)
  }
  __syncthreads();
  float mx = -3.4e38f;
  for (int j = tid; j < 4804; j += 256) mx = fmaxf(mx, lg[j]);
  mx = block_reduce_max(mx, red);
  float se = 0.f;
  for (int j = tid; j < 4804; j += 256) {
    float e = expf(lg[j] - mx); lg[j] = e; se += e;
  }
  float tot = block_reduce_sum(se, red);
  float inv = 1.0f / tot;
  __syncthreads();
  for (int d = tid; d < 1024; d += 256) {
    float acc = 0.f;
    for (int j = 0; j < 4804; ++j)
      acc += lg[j] * KV[((size_t)bi * 4804 + j) * 2048 + 1024 + d];
    O[((size_t)bi * 4 + qi) * 1024 + d] = acc * inv;
  }
}

__global__ __launch_bounds__(256)
void copy_f32(const float* __restrict__ src, float* __restrict__ dst, int n)
{
  int i = blockIdx.x * 256 + threadIdx.x;
  if (i < n) dst[i] = src[i];
}

// out = [ K (4*9 floats) | cls (4*4*1024 floats) ]
__global__ __launch_bounds__(256)
void finalize_k(const float* __restrict__ xsc, const float* __restrict__ S,
                float* __restrict__ out)
{
  int gid = blockIdx.x * 256 + threadIdx.x;
  if (gid < 36) {
    int bi = gid / 9;
    int p  = gid % 9;
    const float* x = xsc + bi * 4;
    float v = 0.f;
    if      (p == 0) v = expf(x[0]);
    else if (p == 4) v = expf(x[1]);
    else if (p == 2) v = 1.f / (1.f + expf(-x[2]));
    else if (p == 5) v = 1.f / (1.f + expf(-x[3]));
    else if (p == 8) v = 1.f;
    out[gid] = v;
  } else if (gid < 36 + 16384) {
    out[gid] = S[gid - 36];
  }
}

// ---------------------------------------------------------------------------
// Host orchestration
// ---------------------------------------------------------------------------
extern "C" void kernel_launch(void* const* d_in, const int* in_sizes, int n_in,
                              void* d_out, int out_size, void* d_ws, size_t ws_size,
                              hipStream_t stream)
{
  (void)in_sizes; (void)n_in; (void)out_size; (void)ws_size;

  const float* features = (const float*)d_in[0];
  const float* cls_tok  = (const float*)d_in[1];
  const float* pos      = (const float*)d_in[2];
  const float* lpos     = (const float*)d_in[3];
  // project_cls (exp=4)
  const float* pc_ng = (const float*)d_in[4];
  const float* pc_nb = (const float*)d_in[5];
  const float* pc_w1 = (const float*)d_in[6];
  const float* pc_b1 = (const float*)d_in[7];
  const float* pc_w2 = (const float*)d_in[8];
  const float* pc_b2 = (const float*)d_in[9];
  // in_features (exp=2)
  const float* inf_ng = (const float*)d_in[10];
  const float* inf_nb = (const float*)d_in[11];
  const float* inf_w1 = (const float*)d_in[12];
  const float* inf_b1 = (const float*)d_in[13];
  const float* inf_w2 = (const float*)d_in[14];
  const float* inf_b2 = (const float*)d_in[15];
  // out head (exp=2, dout=1)
  const float* o_ng = (const float*)d_in[16];
  const float* o_nb = (const float*)d_in[17];
  const float* o_w1 = (const float*)d_in[18];
  const float* o_b1 = (const float*)d_in[19];
  const float* o_w2 = (const float*)d_in[20];
  const float* o_b2 = (const float*)d_in[21];
  // agg1: 22..37, agg2: 38..53 (nxg nxb ncg ncb kv q ow ob ls1 ls2, mlp{ng nb w1 b1 w2 b2})

  char* ws = (char*)d_ws;
  size_t off = 0;
  auto alloc = [&](size_t bytes) -> char* {
    char* p = ws + off;
    off += (bytes + 255) & ~(size_t)255;
    return p;
  };
  bf16*  A0   = (bf16*)alloc((size_t)19216 * 1024 * 2);   // LN'd activations (bf16)
  bf16*  H    = (bf16*)alloc((size_t)19200 * 2048 * 2);   // MLP hidden (bf16)
  float* FEAT = (float*)alloc((size_t)19200 * 1024 * 4);  // in_features output
  float* KV   = (float*)alloc((size_t)19216 * 2048 * 4);  // kv projections
  bf16*  WT1  = (bf16*)alloc((size_t)2048 * 1024 * 2);    // in.w1^T
  bf16*  WT2  = (bf16*)alloc((size_t)1024 * 2048 * 2);    // in.w2^T
  bf16*  WKV1 = (bf16*)alloc((size_t)2048 * 1024 * 2);    // agg1.kv^T
  bf16*  WKV2 = (bf16*)alloc((size_t)2048 * 1024 * 2);    // agg2.kv^T
  float* T0   = (float*)alloc((size_t)16 * 1024 * 4);
  float* T1   = (float*)alloc((size_t)16 * 4096 * 4);
  float* CLS0 = (float*)alloc((size_t)16 * 1024 * 4);
  float* S    = (float*)alloc((size_t)16 * 1024 * 4);
  float* XN   = (float*)alloc((size_t)16 * 1024 * 4);
  float* Q    = (float*)alloc((size_t)16 * 1024 * 4);
  float* ATT  = (float*)alloc((size_t)16 * 1024 * 4);
  float* XSC  = (float*)alloc((size_t)64 * 4);

  const dim3 blk(256);

  // ---- weight prep (bf16, transposed) ----
  transpose_w_bf16<<<dim3(2048/32, 1024/32), blk, 0, stream>>>(inf_w1, WT1, 1024, 2048);
  transpose_w_bf16<<<dim3(1024/32, 2048/32), blk, 0, stream>>>(inf_w2, WT2, 2048, 1024);
  transpose_w_bf16<<<dim3(2048/32, 1024/32), blk, 0, stream>>>((const float*)d_in[26], WKV1, 1024, 2048);
  transpose_w_bf16<<<dim3(2048/32, 1024/32), blk, 0, stream>>>((const float*)d_in[42], WKV2, 1024, 2048);

  // ---- feats = MLP(moveaxis(features)+pos)  (two WMMA GEMMs) ----
  prep_in_features<<<19200, blk, 0, stream>>>(features, pos, inf_ng, inf_nb, A0);
  wmma_gemm_bf16<<<dim3(2048/128, 19200/128), blk, 0, stream>>>(
      A0, WT1, inf_b1, H, 19200, 2048, 1024, /*gelu|bf16out*/ 3);
  wmma_gemm_bf16<<<dim3(1024/128, 19200/128), blk, 0, stream>>>(
      H, WT2, inf_b2, FEAT, 19200, 1024, 2048, 0);

  // ---- cls = project_cls MLP (16 rows) ----
  ln_rows<<<16, blk, 0, stream>>>(cls_tok, pc_ng, pc_nb, T0);
  fc_small<<<16, blk, 0, stream>>>(T0, pc_w1, pc_b1, T1, 1024, 4096, 1, nullptr, nullptr, nullptr);
  fc_small<<<16, blk, 0, stream>>>(T1, pc_w2, pc_b2, CLS0, 4096, 1024, 0, nullptr, nullptr, nullptr);
  copy_f32<<<64, blk, 0, stream>>>(CLS0, S, 16384);

  // ---- two attention blocks (ctx = [feats | cls0] is fixed) ----
  for (int ab = 0; ab < 2; ++ab) {
    const int base = 22 + ab * 16;
    const float* nxg  = (const float*)d_in[base + 0];
    const float* nxb  = (const float*)d_in[base + 1];
    const float* ncg  = (const float*)d_in[base + 2];
    const float* ncb  = (const float*)d_in[base + 3];
    const float* Wq   = (const float*)d_in[base + 5];
    const float* Wow  = (const float*)d_in[base + 6];
    const float* ob   = (const float*)d_in[base + 7];
    const float* ls1  = (const float*)d_in[base + 8];
    const float* ls2  = (const float*)d_in[base + 9];
    const float* m_ng = (const float*)d_in[base + 10];
    const float* m_nb = (const float*)d_in[base + 11];
    const float* m_w1 = (const float*)d_in[base + 12];
    const float* m_b1 = (const float*)d_in[base + 13];
    const float* m_w2 = (const float*)d_in[base + 14];
    const float* m_b2 = (const float*)d_in[base + 15];
    bf16* WKV = ab ? WKV2 : WKV1;

    prep_ctx<<<19216, blk, 0, stream>>>(FEAT, CLS0, ncg, ncb, A0);
    wmma_gemm_bf16<<<dim3(2048/128, (19216 + 127)/128), blk, 0, stream>>>(
        A0, WKV, nullptr, KV, 19216, 2048, 1024, 0);

    ln_rows<<<16, blk, 0, stream>>>(S, nxg, nxb, XN);
    fc_small<<<16, blk, 0, stream>>>(XN, Wq, nullptr, Q, 1024, 1024, 0, lpos, nullptr, nullptr);
    attn_small<<<16, blk, 0, stream>>>(Q, KV, ATT);
    fc_small<<<16, blk, 0, stream>>>(ATT, Wow, ob, S, 1024, 1024, 0, nullptr, S, ls1);
    ln_rows<<<16, blk, 0, stream>>>(S, m_ng, m_nb, XN);
    fc_small<<<16, blk, 0, stream>>>(XN, m_w1, m_b1, T1, 1024, 4096, 1, nullptr, nullptr, nullptr);
    fc_small<<<16, blk, 0, stream>>>(T1, m_w2, m_b2, S, 4096, 1024, 0, nullptr, S, ls2);
  }

  // ---- out head + K matrix ----
  ln_rows<<<16, blk, 0, stream>>>(S, o_ng, o_nb, XN);
  fc_small<<<16, blk, 0, stream>>>(XN, o_w1, o_b1, T1, 1024, 2048, 1, nullptr, nullptr, nullptr);
  fc_small<<<16, blk, 0, stream>>>(T1, o_w2, o_b2, XSC, 2048, 1, 0, nullptr, nullptr, nullptr);
  finalize_k<<<65, blk, 0, stream>>>(XSC, S, (float*)d_out);
}
